// Model_51436528337215
// MI455X (gfx1250) — matrix operand order repaired and checked by reading, not compiled, not run on hardware
//
#include <hip/hip_runtime.h>

typedef _Float16 v16h __attribute__((ext_vector_type(16)));
typedef __fp16   v2fp __attribute__((ext_vector_type(2)));   // cvt_pkrtz result
typedef float    v8f  __attribute__((ext_vector_type(8)));
typedef unsigned int u32x4 __attribute__((ext_vector_type(4)));
typedef int          i32x4 __attribute__((ext_vector_type(4)));
typedef int          i32x8 __attribute__((ext_vector_type(8)));

#define HNUM 3
#define NSQ  49
#define HD   32
#define CCH  96
#define IMGL (56*56*96)   // elements per image
#define WPB  4            // windows per block
#define BLK  384          // 12 waves: wave wv -> (window wv/3, head wv%3)
#define PRS  100          // dword stride per row-pair (96 ch + 4 pad)

// LDS layout (elements):
//   xs  : float    [4][49][96]        raw window data (residual)   75264 B
//   xn2 : _Float16 [4][32][100][2]    LN'd f16, row-pair packed:   51200 B
//         dword (win,p,c) = {row 2p ch c, row 2p+1 ch c}; rows>=49 zero
//   wA  : _Float16 [3][64][66]        zero-padded weights          25344 B
//   sg/sbe : float[96] each; sb : float[147]                        1356 B
#define SMEM_BYTES (WPB*NSQ*CCH*4 + WPB*32*PRS*4 + 3*64*66*2 + 2*CCH*4 + HNUM*NSQ*4)

__global__ __launch_bounds__(BLK) void swinmlp_wmma_kernel(
    const float* __restrict__ x, const float* __restrict__ gamma,
    const float* __restrict__ beta, const float* __restrict__ w,
    const float* __restrict__ bias, float* __restrict__ out)
{
    extern __shared__ __align__(16) unsigned char smem_raw[];
    float*    xs  = (float*)smem_raw;                  // WPB*49*96 f32
    _Float16* xn2 = (_Float16*)(xs + WPB*NSQ*CCH);     // WPB*32*PRS dwords
    _Float16* wA  = xn2 + WPB*32*PRS*2;                // 3*64*66 f16
    float*    sg  = (float*)(wA + 3*64*66);            // 96
    float*    sbe = sg + CCH;                          // 96
    float*    sb  = sbe + CCH;                         // 147

    const int tid  = threadIdx.x;
    const int lane = tid & 31;
    const int wv   = tid >> 5;          // 0..11 (wave32)
    const int n16  = lane & 15;
    const int hi   = lane >> 4;         // 0 / 1 (half-wave)

    // ---- stage gamma/beta/bias ----
    if (tid < CCH) { sg[tid] = gamma[tid]; sbe[tid] = beta[tid]; }
    if (tid < HNUM*NSQ) sb[tid] = bias[tid];

    // ---- build zero-padded f16 weight image wA[h][64][66] ----
    for (int idx = tid; idx < 3*64*66; idx += BLK) {
        int h = idx / (64*66); int rem = idx - h*(64*66);
        int i = rem / 66, j = rem - i*66;
        _Float16 v = (_Float16)0.0f;
        if (i < NSQ && j < NSQ) v = (_Float16)w[(h*NSQ + i)*NSQ + j];
        wA[idx] = v;
    }

    // ---- TDM: DMA all 4 windows' tiles into LDS (wave 0 issues) ----
    if (wv == 0) {
#pragma unroll
        for (int win = 0; win < WPB; ++win) {
            int gwin = blockIdx.x*WPB + win;
            int bimg = gwin >> 6;
            int widx = gwin & 63;
            int wr = widx >> 3, wc = widx & 7;
            unsigned long long gaddr = (unsigned long long)(const void*)x
                + 4ull * ((unsigned long long)bimg*IMGL
                          + (unsigned long long)((wr*7)*56 + wc*7)*CCH);
            unsigned int laddr =
                (unsigned int)(unsigned long long)(void*)(xs + win*NSQ*CCH);
            // D# group0: count=1, lds_addr, global_addr, type=2
            u32x4 g0;
            g0[0] = 1u;
            g0[1] = laddr;
            g0[2] = (unsigned int)(gaddr & 0xFFFFFFFFull);
            g0[3] = (unsigned int)((gaddr >> 32) & 0x1FFFFFFull) | (2u << 30);
            // D# group1: data_size=4B(code 2); 2D tile: dim0=672 contiguous
            // floats (7 cols x 96 ch), dim1=7 rows with stride 56*96
            i32x8 g1;
            g1[0] = (int)(2u << 16);        // data_size
            g1[1] = (int)(672u << 16);      // tensor_dim0[15:0]
            g1[2] = (int)(7u << 16);        // tensor_dim1[15:0] (dim0 hi = 0)
            g1[3] = (int)(672u << 16);      // tile_dim0 (dim1 hi = 0)
            g1[4] = 7;                      // tile_dim1=7, tile_dim2=0
            g1[5] = 56*CCH;                 // tensor_dim0_stride lo
            g1[6] = 0;
            g1[7] = 0;
            i32x4 g2 = {0,0,0,0}, g3 = {0,0,0,0};
#if defined(__clang_major__) && (__clang_major__ >= 23)
            i32x8 g4 = {0,0,0,0,0,0,0,0};
            __builtin_amdgcn_tensor_load_to_lds(g0, g1, g2, g3, g4, 0);
#else
            __builtin_amdgcn_tensor_load_to_lds(g0, g1, g2, g3, 0);
#endif
        }
        __builtin_amdgcn_s_wait_tensorcnt(0);
    }
    __syncthreads();

    // ---- LayerNorm: one thread per ROW PAIR (WPB*32 pairs), f32 -> f16 ----
    // Pad rows (>=49) are zeroed by masking the packed dwords; no per-element
    // multiplies, no divergent branches, no OOB reads (clamped row pointers).
    {
        const float4* g4  = (const float4*)sg;
        const float4* be4 = (const float4*)sbe;
        for (int t = tid; t < WPB*32; t += BLK) {
            int win = t >> 5, p = t & 31;
            int ka = 2*p, kb = ka + 1;
            unsigned int mask = (ka < NSQ ? 0x0000FFFFu : 0u)
                              | (kb < NSQ ? 0xFFFF0000u : 0u);
            int kaC = (ka < NSQ) ? ka : (NSQ-1);
            int kbC = (kb < NSQ) ? kb : (NSQ-1);
            const float4* ra = (const float4*)(xs + (win*NSQ + kaC)*CCH);
            const float4* rb = (const float4*)(xs + (win*NSQ + kbC)*CCH);
            float sa = 0.f, ssa = 0.f, sbm = 0.f, ssb = 0.f;
#pragma unroll
            for (int k = 0; k < CCH/4; ++k) {
                float4 a = ra[k], b = rb[k];
                sa  += a.x + a.y + a.z + a.w;
                ssa += a.x*a.x + a.y*a.y + a.z*a.z + a.w*a.w;
                sbm += b.x + b.y + b.z + b.w;
                ssb += b.x*b.x + b.y*b.y + b.z*b.z + b.w*b.w;
            }
            float mua = sa  * (1.0f/CCH);
            float mub = sbm * (1.0f/CCH);
            float rsa = rsqrtf(ssa*(1.0f/CCH) - mua*mua + 1e-5f);
            float rsb = rsqrtf(ssb*(1.0f/CCH) - mub*mub + 1e-5f);
            uint4* orow = (uint4*)(xn2 + (win*32 + p)*PRS*2);
#pragma unroll
            for (int k = 0; k < CCH/4; ++k) {
                float4 a = ra[k], b = rb[k];
                float4 g = g4[k], be = be4[k];
                union { v2fp h; unsigned int u; } q0, q1, q2, q3;
                q0.h = __builtin_amdgcn_cvt_pkrtz((a.x-mua)*rsa*g.x + be.x,
                                                  (b.x-mub)*rsb*g.x + be.x);
                q1.h = __builtin_amdgcn_cvt_pkrtz((a.y-mua)*rsa*g.y + be.y,
                                                  (b.y-mub)*rsb*g.y + be.y);
                q2.h = __builtin_amdgcn_cvt_pkrtz((a.z-mua)*rsa*g.z + be.z,
                                                  (b.z-mub)*rsb*g.z + be.z);
                q3.h = __builtin_amdgcn_cvt_pkrtz((a.w-mua)*rsa*g.w + be.w,
                                                  (b.w-mub)*rsb*g.w + be.w);
                uint4 o;
                o.x = q0.u & mask; o.y = q1.u & mask;
                o.z = q2.u & mask; o.w = q3.u & mask;
                orow[k] = o;
            }
        }
    }
    __syncthreads();

    // ---- per-wave matmul: Y(49x32) = W_h(49x49) @ X_h(49x32) ----
    const int win  = wv / HNUM;         // 0..3
    const int h    = wv - win*HNUM;     // 0..2
    const int gwin = blockIdx.x*WPB + win;
    const int bimg = gwin >> 6;
    const int widx = gwin & 63;
    const int wr = widx >> 3, wc = widx & 7;
    const _Float16* xnw = xn2 + win*32*PRS*2;

    v8f acc[4][2];
#pragma unroll
    for (int mt = 0; mt < 4; ++mt)
#pragma unroll
        for (int nt = 0; nt < 2; ++nt)
            acc[mt][nt] = (v8f){0.f,0.f,0.f,0.f,0.f,0.f,0.f,0.f};

#pragma unroll
    for (int ks = 0; ks < 2; ++ks) {
        // B fragments (32x16 f16): lane n16 = column N;
        // dword r = rows {K, K+1} with K = ks*32 + hi*16 + 2r -> pair ks*16+hi*8+r
        union { v16h v; unsigned int u[8]; } B[2];
#pragma unroll
        for (int nt = 0; nt < 2; ++nt) {
            int c = h*HD + nt*16 + n16;
#pragma unroll
            for (int r = 0; r < 8; ++r) {
                int pr = ks*16 + hi*8 + r;
                B[nt].u[r] = *(const unsigned int*)(xnw + (pr*PRS + c)*2);
            }
        }
#pragma unroll
        for (int mt = 0; mt < 4; ++mt) {
            // A fragment (16x32 f16): lane n16 = row M;
            // K = hi*16 + r*2 (+32*ks), packed f16 pairs
            union { v16h v; unsigned int u[8]; } A;
            const _Float16* wrow = wA + (h*64 + mt*16 + n16)*66;
#pragma unroll
            for (int r = 0; r < 8; ++r) {
                int k = ks*32 + hi*16 + r*2;
                A.u[r] = *(const unsigned int*)(wrow + k);
            }
#pragma unroll
            for (int nt = 0; nt < 2; ++nt) {
                acc[mt][nt] = __builtin_amdgcn_wmma_f32_16x16x32_f16(
                    false, A.v, false, B[nt].v, (short)0, acc[mt][nt], false, false);
            }
        }
    }

    // ---- fused bias + residual + window-reverse store ----
    const size_t obase = (size_t)bimg * IMGL;
#pragma unroll
    for (int mt = 0; mt < 4; ++mt) {
#pragma unroll
        for (int nt = 0; nt < 2; ++nt) {
#pragma unroll
            for (int r = 0; r < 8; ++r) {
                int i = mt*16 + hi*8 + r;            // D: M = mt*16 + hi*8 + r
                if (i < NSQ) {
                    int c = h*HD + nt*16 + n16;      // D: N = lane&15
                    float v = acc[mt][nt][r]
                            + xs[(win*NSQ + i)*CCH + c]
                            + sb[h*NSQ + i];
                    int rr = i / 7, cc = i - rr*7;
                    size_t idx = obase
                        + (size_t)((wr*7 + rr)*56 + (wc*7 + cc))*CCH + c;
                    out[idx] = v;
                }
            }
        }
    }
}

extern "C" void kernel_launch(void* const* d_in, const int* in_sizes, int n_in,
                              void* d_out, int out_size, void* d_ws, size_t ws_size,
                              hipStream_t stream) {
    const float* x     = (const float*)d_in[0];
    const float* gamma = (const float*)d_in[1];
    const float* beta  = (const float*)d_in[2];
    const float* w     = (const float*)d_in[3];
    const float* b     = (const float*)d_in[4];
    float* out = (float*)d_out;

    int n_img   = in_sizes[0] / IMGL;      // 256
    int n_win   = n_img * 64;              // 16384 windows
    int n_blk   = n_win / WPB;             // 4 windows per block

    dim3 grid(n_blk), block(BLK);
    hipLaunchKernelGGL(swinmlp_wmma_kernel, grid, block, SMEM_BYTES, stream,
                       x, gamma, beta, w, b, out);
}
